// M_GCN_t_22969485099901
// MI455X (gfx1250) — matrix operand order repaired and checked by reading, not compiled, not run on hardware
//
#include <hip/hip_runtime.h>
#include <hip/hip_bf16.h>
#include <stdint.h>

// ---------------------------------------------------------------------------
// M-GCN forward for MI455X (gfx1250, wave32, WMMA).
// All dense GEMMs run on v_wmma_f32_16x16x32_f16 (f16 in, f32 accumulate).
// Irregular graph ops (segment softmax / scatter-add) use float atomics.
// ---------------------------------------------------------------------------

typedef _Float16 f16;
typedef __attribute__((ext_vector_type(16))) _Float16 v16h;
typedef __attribute__((ext_vector_type(8)))  float    v8f;

#define NN    20000
#define EE    320000
#define HH    128
#define VV    3
#define TT    (VV * NN)     // 60000 tokens
#define FEAT  256
#define FFH   512
#define NEGSL 0.2f

// ---------------------------------------------------------------------------
// WMMA GEMM:  C[M,Nc] (f32 and/or f16) = act( A[M,K](f16,row-major) *
//             Bt[Nc,K](f16, pre-transposed weight) + bias )
//
// Each wave computes a 16x64 strip: ONE A fragment is reused by FOUR B
// fragments / accumulators per K-step (4 WMMAs per A load).
// Two-stage software pipeline: fragments for K-step i+1 are loaded while
// WMMAs for K-step i execute, so waits can be partial (s_wait_loadcnt N>0)
// instead of full drains. K/32 is even for all K in {128,256,512} -> no tail.
// Block = 4 waves = 2 M-tiles x 2 col-groups = 32 x 128 per block.
// grid = (M/32, Nc/128); M in {20000,60000}, Nc in {128,512} -> exact.
//
// Fragment layouts as loaded here (wave32; k contiguous per lane half, not the ISA table order):
//   A 16x32 f16 : lanes 0-15 -> M=lane, K=0..15;
//                 lanes 16-31 -> M=lane-16, K=16..31
//   B 32x16 f16 : lanes 0-15 -> N=lane, K=0..15; lanes 16-31 -> K=16..31 (same order as A)
//   C/D 16x16 f32: vgpr r holds M=r (lanes 0-15) / M=8+r (lanes 16-31), N=lane&15
// ---------------------------------------------------------------------------
union Frag { v16h h; uint32_t u[8]; };

template <int ACT>   // 0=none 1=relu 2=tanh
__global__ __launch_bounds__(128)
void gemm_wmma_f16(const f16* __restrict__ A, const f16* __restrict__ Bt,
                   const float* __restrict__ bias,
                   float* __restrict__ C, f16* __restrict__ C16,
                   int M, int Nc, int K) {
  const int wave  = threadIdx.x >> 5;
  const int lane  = threadIdx.x & 31;
  const int tileM = blockIdx.x * 32 + (wave & 1) * 16;
  const int colG  = blockIdx.y * 128 + (wave >> 1) * 64;   // 64-wide strip
  if (tileM >= M || colG >= Nc) return;   // uniform per wave (never taken for our sizes)

  const int  l15 = lane & 15;
  const int  hi  = lane >> 4;             // 0 or 1
  const long long arow  = (long long)(tileM + l15) * K;
  const long long brow0 = (long long)(colG +  0 + l15) * K;
  const long long brow1 = (long long)(colG + 16 + l15) * K;
  const long long brow2 = (long long)(colG + 32 + l15) * K;
  const long long brow3 = (long long)(colG + 48 + l15) * K;

  v8f acc0 = {}, acc1 = {}, acc2 = {}, acc3 = {};

  // load one K-step's fragments (A + 4 B strips) at column offset k0
  auto LD = [&](Frag& af, Frag& bf0, Frag& bf1, Frag& bf2, Frag& bf3, int k0) {
    const uint32_t* Ap  = reinterpret_cast<const uint32_t*>(A  + arow  + k0);
    const uint32_t* Bp0 = reinterpret_cast<const uint32_t*>(Bt + brow0 + k0);
    const uint32_t* Bp1 = reinterpret_cast<const uint32_t*>(Bt + brow1 + k0);
    const uint32_t* Bp2 = reinterpret_cast<const uint32_t*>(Bt + brow2 + k0);
    const uint32_t* Bp3 = reinterpret_cast<const uint32_t*>(Bt + brow3 + k0);
#pragma unroll
    for (int j = 0; j < 8; ++j) af.u[j]     = Ap[hi * 8 + j];          // K = 16*hi + 2j
#pragma unroll
    for (int j = 0; j < 8; ++j) bf0.u[j] = Bp0[hi * 8 + j];            // K = 16*hi + 2j
#pragma unroll
    for (int j = 0; j < 8; ++j) bf1.u[j] = Bp1[hi * 8 + j];
#pragma unroll
    for (int j = 0; j < 8; ++j) bf2.u[j] = Bp2[hi * 8 + j];
#pragma unroll
    for (int j = 0; j < 8; ++j) bf3.u[j] = Bp3[hi * 8 + j];
  };
  auto MM = [&](Frag& af, Frag& bf0, Frag& bf1, Frag& bf2, Frag& bf3) {
    acc0 = __builtin_amdgcn_wmma_f32_16x16x32_f16(false, af.h, false, bf0.h,
                                                  (short)0, acc0, false, false);
    acc1 = __builtin_amdgcn_wmma_f32_16x16x32_f16(false, af.h, false, bf1.h,
                                                  (short)0, acc1, false, false);
    acc2 = __builtin_amdgcn_wmma_f32_16x16x32_f16(false, af.h, false, bf2.h,
                                                  (short)0, acc2, false, false);
    acc3 = __builtin_amdgcn_wmma_f32_16x16x32_f16(false, af.h, false, bf3.h,
                                                  (short)0, acc3, false, false);
  };

  Frag aA, bA0, bA1, bA2, bA3;    // pipeline stage 0
  Frag aB, bB0, bB1, bB2, bB3;    // pipeline stage 1

  const int steps = K >> 5;       // 4, 8 or 16 (always even)
  LD(aA, bA0, bA1, bA2, bA3, 0);
  LD(aB, bB0, bB1, bB2, bB3, 32);
  int k0 = 64;
  for (int it = 0; it < (steps >> 1) - 1; ++it) {
    MM(aA, bA0, bA1, bA2, bA3);
    LD(aA, bA0, bA1, bA2, bA3, k0);
    MM(aB, bB0, bB1, bB2, bB3);
    LD(aB, bB0, bB1, bB2, bB3, k0 + 32);
    k0 += 64;
  }
  MM(aA, bA0, bA1, bA2, bA3);
  MM(aB, bB0, bB1, bB2, bB3);

  const int mb = tileM + hi * 8;
  v8f accs[4] = { acc0, acc1, acc2, acc3 };
#pragma unroll
  for (int t = 0; t < 4; ++t) {
    const int   n  = colG + t * 16 + l15;
    const float bv = bias ? bias[n] : 0.0f;
#pragma unroll
    for (int r = 0; r < 8; ++r) {
      float v = accs[t][r] + bv;
      if (ACT == 1) v = v > 0.0f ? v : 0.0f;
      else if (ACT == 2) v = tanhf(v);
      const long long idx = (long long)(mb + r) * Nc + n;
      if (C)   C[idx]   = v;
      if (C16) C16[idx] = (f16)v;
    }
  }
}

// ---------------------------------------------------------------------------
// Small utility kernels
// ---------------------------------------------------------------------------
__global__ void k_zero_u32(uint32_t* __restrict__ p, long long n) {
  long long i = (long long)blockIdx.x * blockDim.x + threadIdx.x;
  long long s = (long long)gridDim.x * blockDim.x;
  for (; i < n; i += s) p[i] = 0u;
}

__global__ void k_f32_to_f16(const float* __restrict__ in, f16* __restrict__ out,
                             long long n) {
  long long i = (long long)blockIdx.x * blockDim.x + threadIdx.x;
  long long s = (long long)gridDim.x * blockDim.x;
  for (; i < n; i += s) out[i] = (f16)in[i];
}

// W [K,Nc] f32 row-major -> Wt [Nc,K] f16 (so GEMM B-fragment loads are contiguous)
__global__ void k_w_transpose_f16(const float* __restrict__ W, f16* __restrict__ Wt,
                                  int K, int Nc) {
  long long idx = (long long)blockIdx.x * blockDim.x + threadIdx.x;
  if (idx >= (long long)K * Nc) return;
  int kk = (int)(idx / Nc), nn = (int)(idx % Nc);
  Wt[(long long)nn * K + kk] = (f16)W[idx];
}

// Two simultaneous 128-dim row dots (wave per row, wave32 shuffle reduce)
__global__ __launch_bounds__(256)
void k_rowdot2(const float* __restrict__ X, const float* __restrict__ a0,
               const float* __restrict__ a1, float* __restrict__ o0,
               float* __restrict__ o1, int nrows) {
  int row = blockIdx.x * 8 + (threadIdx.x >> 5);
  if (row >= nrows) return;
  int lane = threadIdx.x & 31;
  const float* xr = X + (long long)row * HH;
  float s0 = 0.f, s1 = 0.f;
#pragma unroll
  for (int i = 0; i < 4; ++i) {
    int c = lane + 32 * i;
    float x = xr[c];
    s0 += x * a0[c];
    if (a1) s1 += x * a1[c];
  }
#pragma unroll
  for (int off = 16; off; off >>= 1) {
    s0 += __shfl_xor(s0, off, 32);
    s1 += __shfl_xor(s1, off, 32);
  }
  if (lane == 0) { o0[row] = s0; if (o1) o1[row] = s1; }
}

// ---------------------------------------------------------------------------
// GAT edge kernels: segment-max / exp+denom / scatter-aggregate
// (edges e < E come from edge_index, e >= E are the N self loops)
// Float atomic-max via order-preserving uint key -> GLOBAL_ATOMIC_MAX_U32.
// ---------------------------------------------------------------------------
__device__ __forceinline__ unsigned fkey(float f) {
  unsigned u = __float_as_uint(f);
  return (u & 0x80000000u) ? ~u : (u | 0x80000000u);
}
__device__ __forceinline__ float funkey(unsigned k) {
  return __uint_as_float((k & 0x80000000u) ? (k ^ 0x80000000u) : ~k);
}
__device__ __forceinline__ float lrelu(float v) { return v > 0.f ? v : NEGSL * v; }

__global__ void k_edge_max(const int* __restrict__ src, const int* __restrict__ dst,
                           const float* __restrict__ as, const float* __restrict__ ad,
                           unsigned* __restrict__ maxb) {
  int e = blockIdx.x * 256 + threadIdx.x;
  if (e >= EE + NN) return;
  int s, d;
  if (e < EE) { s = src[e]; d = dst[e]; } else { s = d = e - EE; }
  atomicMax(&maxb[d], fkey(lrelu(as[s] + ad[d])));
}

__global__ void k_edge_exp(const int* __restrict__ src, const int* __restrict__ dst,
                           const float* __restrict__ as, const float* __restrict__ ad,
                           const unsigned* __restrict__ maxb,
                           float* __restrict__ eexp, float* __restrict__ denom) {
  int e = blockIdx.x * 256 + threadIdx.x;
  if (e >= EE + NN) return;
  int s, d;
  if (e < EE) { s = src[e]; d = dst[e]; } else { s = d = e - EE; }
  float ex = __expf(lrelu(as[s] + ad[d]) - funkey(maxb[d]));
  eexp[e] = ex;
  atomicAdd(&denom[d], ex);
}

// wave per edge, 4 columns per lane -> coalesced GLOBAL_ATOMIC_ADD_F32
__global__ __launch_bounds__(256)
void k_edge_agg(const int* __restrict__ src, const int* __restrict__ dst,
                const float* __restrict__ eexp, const float* __restrict__ denom,
                const float* __restrict__ xp, float* __restrict__ agg) {
  int e = blockIdx.x * 8 + (threadIdx.x >> 5);
  if (e >= EE + NN) return;
  int lane = threadIdx.x & 31;
  int s, d;
  if (e < EE) { s = src[e]; d = dst[e]; } else { s = d = e - EE; }
  float alpha = eexp[e] / (denom[d] + 1e-16f);
  const float* xs = xp + (long long)s * HH;
  float*       od = agg + (long long)d * HH;
#pragma unroll
  for (int i = 0; i < 4; ++i) {
    int c = lane + 32 * i;
    atomicAdd(&od[c], alpha * xs[c]);
  }
}

__global__ void k_gat_fin(const float* __restrict__ agg, const float* __restrict__ b,
                          float* __restrict__ outf, f16* __restrict__ out16) {
  long long idx = (long long)blockIdx.x * blockDim.x + threadIdx.x;
  if (idx >= (long long)NN * HH) return;
  int c = (int)(idx & (HH - 1));
  float v = agg[idx] + b[c];
  v = v > 0.f ? v : 0.f;                 // relu from _mv_gnn
  outf[idx] = v;
  out16[idx] = (f16)v;
}

// ---------------------------------------------------------------------------
// Encoder: tiny S=3, heads=8, hd=16 attention. One thread per (batch,head).
// ---------------------------------------------------------------------------
__global__ __launch_bounds__(256)
void k_attn(const float* __restrict__ q, const float* __restrict__ k,
            const float* __restrict__ v, f16* __restrict__ o16) {
  int t = blockIdx.x * 256 + threadIdx.x;
  if (t >= NN * 8) return;
  int b = t >> 3, h = t & 7;
  long long base = (long long)b * HH + h * 16;
  const long long strS = (long long)NN * HH;

  float qv[3][16];
#pragma unroll
  for (int s = 0; s < 3; ++s)
#pragma unroll
    for (int d = 0; d < 16; ++d) qv[s][d] = q[base + s * strS + d];

  float sc[3][3];
#pragma unroll
  for (int tt = 0; tt < 3; ++tt) {
    float kv[16];
#pragma unroll
    for (int d = 0; d < 16; ++d) kv[d] = k[base + tt * strS + d];
#pragma unroll
    for (int s = 0; s < 3; ++s) {
      float a = 0.f;
#pragma unroll
      for (int d = 0; d < 16; ++d) a += qv[s][d] * kv[d];
      sc[s][tt] = a * 0.25f;             // 1/sqrt(16)
    }
  }
  float aw[3][3];
#pragma unroll
  for (int s = 0; s < 3; ++s) {
    float mx = fmaxf(sc[s][0], fmaxf(sc[s][1], sc[s][2]));
    float e0 = __expf(sc[s][0] - mx), e1 = __expf(sc[s][1] - mx), e2 = __expf(sc[s][2] - mx);
    float inv = 1.f / (e0 + e1 + e2);
    aw[s][0] = e0 * inv; aw[s][1] = e1 * inv; aw[s][2] = e2 * inv;
  }
#pragma unroll
  for (int d = 0; d < 16; ++d) {
    float v0 = v[base + 0 * strS + d], v1 = v[base + 1 * strS + d], v2 = v[base + 2 * strS + d];
#pragma unroll
    for (int s = 0; s < 3; ++s)
      o16[base + s * strS + d] = (f16)(aw[s][0] * v0 + aw[s][1] * v1 + aw[s][2] * v2);
  }
}

// LayerNorm(x + r) over 128 features; wave per row
__global__ __launch_bounds__(256)
void k_ln_res(const float* __restrict__ x, const float* __restrict__ r,
              const float* __restrict__ w, const float* __restrict__ b,
              float* __restrict__ outf, f16* __restrict__ out16, int nrows) {
  int row = blockIdx.x * 8 + (threadIdx.x >> 5);
  if (row >= nrows) return;
  int lane = threadIdx.x & 31;
  long long base = (long long)row * HH;
  float vals[4]; float s = 0.f;
#pragma unroll
  for (int i = 0; i < 4; ++i) {
    int c = lane + 32 * i;
    float t = x[base + c] + (r ? r[base + c] : 0.f);
    vals[i] = t; s += t;
  }
#pragma unroll
  for (int off = 16; off; off >>= 1) s += __shfl_xor(s, off, 32);
  float mu = s * (1.f / HH);
  float vv = 0.f;
#pragma unroll
  for (int i = 0; i < 4; ++i) { float d = vals[i] - mu; vv += d * d; }
#pragma unroll
  for (int off = 16; off; off >>= 1) vv += __shfl_xor(vv, off, 32);
  float inv = rsqrtf(vv * (1.f / HH) + 1e-5f);
#pragma unroll
  for (int i = 0; i < 4; ++i) {
    int c = lane + 32 * i;
    float t = (vals[i] - mu) * inv * w[c] + b[c];
    outf[base + c] = t;
    if (out16) out16[base + c] = (f16)t;
  }
}

// beta = softmax over views of w; g = sum_v beta_v * a2_v
__global__ void k_pool(const float* __restrict__ wv, const float* __restrict__ a2,
                       float* __restrict__ g) {
  long long idx = (long long)blockIdx.x * blockDim.x + threadIdx.x;
  if (idx >= (long long)NN * HH) return;
  int n = (int)(idx >> 7);
  float w0 = wv[n], w1 = wv[NN + n], w2 = wv[2 * NN + n];
  float mx = fmaxf(w0, fmaxf(w1, w2));
  float e0 = __expf(w0 - mx), e1 = __expf(w1 - mx), e2 = __expf(w2 - mx);
  float inv = 1.f / (e0 + e1 + e2);
  const long long S = (long long)NN * HH;
  g[idx] = (e0 * a2[idx] + e1 * a2[S + idx] + e2 * a2[2 * S + idx]) * inv;
}

// ---------------------------------------------------------------------------
// Host orchestration
// ---------------------------------------------------------------------------
struct EncDev {
  f16 *Wqt, *Wkt, *Wvt, *Wot, *W1t, *W2t;
  const float *bq, *bk, *bv, *bo, *b1, *b2, *ln1w, *ln1b, *ln2w, *ln2b;
};

extern "C" void kernel_launch(void* const* d_in, const int* in_sizes, int n_in,
                              void* d_out, int out_size, void* d_ws, size_t ws_size,
                              hipStream_t stream) {
  (void)in_sizes; (void)n_in; (void)out_size; (void)ws_size;
  // ---- input mapping: setup_inputs() dict order, params JAX-tree (sorted keys)
  const float* xin[VV] = { (const float*)d_in[0], (const float*)d_in[2], (const float*)d_in[4] };
  const int*   ei[VV]  = { (const int*)d_in[1],   (const int*)d_in[3],   (const int*)d_in[5]   };
  int p = 6;
  auto F = [&](int i) { return (const float*)d_in[i]; };
  // att: W1, b1, w2
  const float *attW1 = F(p), *attb1 = F(p + 1), *attw2 = F(p + 2); p += 3;
  // enc: W1, W2, Wk, Wo, Wq, Wv, b1, b2, bk, bo, bq, bv, ln1_b, ln1_w, ln2_b, ln2_w
  const float* encp[2][16];
  for (int e = 0; e < 2; ++e) { for (int i = 0; i < 16; ++i) encp[e][i] = F(p + i); p += 16; }
  // gat: (W, a_dst, a_src, b) x3, for gat1 then gat2
  const float *g1W[VV], *g1ad[VV], *g1as[VV], *g1b[VV];
  const float *g2W[VV], *g2ad[VV], *g2as[VV], *g2b[VV];
  for (int i = 0; i < VV; ++i) { g1W[i]=F(p); g1ad[i]=F(p+1); g1as[i]=F(p+2); g1b[i]=F(p+3); p+=4; }
  for (int i = 0; i < VV; ++i) { g2W[i]=F(p); g2ad[i]=F(p+1); g2as[i]=F(p+2); g2b[i]=F(p+3); p+=4; }

  // ---- workspace bump allocator
  size_t off = 0;
  auto alloc = [&](size_t bytes) -> void* {
    void* r = (char*)d_ws + off;
    off += (bytes + 255) & ~(size_t)255;
    return r;
  };
  f16* g1Wt[VV]; for (int i = 0; i < VV; ++i) g1Wt[i] = (f16*)alloc((size_t)HH * FEAT * 2);
  f16* g2Wt[VV]; for (int i = 0; i < VV; ++i) g2Wt[i] = (f16*)alloc((size_t)HH * HH * 2);
  EncDev enc[2];
  for (int e = 0; e < 2; ++e) {
    enc[e].Wqt = (f16*)alloc((size_t)HH * HH * 2);
    enc[e].Wkt = (f16*)alloc((size_t)HH * HH * 2);
    enc[e].Wvt = (f16*)alloc((size_t)HH * HH * 2);
    enc[e].Wot = (f16*)alloc((size_t)HH * HH * 2);
    enc[e].W1t = (f16*)alloc((size_t)FFH * HH * 2);
    enc[e].W2t = (f16*)alloc((size_t)HH * FFH * 2);
    enc[e].b1 = encp[e][6];  enc[e].b2 = encp[e][7];
    enc[e].bk = encp[e][8];  enc[e].bo = encp[e][9];
    enc[e].bq = encp[e][10]; enc[e].bv = encp[e][11];
    enc[e].ln1b = encp[e][12]; enc[e].ln1w = encp[e][13];
    enc[e].ln2b = encp[e][14]; enc[e].ln2w = encp[e][15];
  }
  f16* attW1t = (f16*)alloc((size_t)HH * HH * 2);
  f16* x16[VV]; for (int i = 0; i < VV; ++i) x16[i] = (f16*)alloc((size_t)NN * FEAT * 2);
  float*    xp     = (float*)alloc((size_t)TT * HH * 4);
  float*    a_s    = (float*)alloc((size_t)NN * 4);
  float*    a_d    = (float*)alloc((size_t)NN * 4);
  unsigned* maxb   = (unsigned*)alloc((size_t)NN * 4);
  float*    denom  = (float*)alloc((size_t)NN * 4);
  float*    eexp   = (float*)alloc((size_t)(EE + NN) * 4);
  float*    agg    = (float*)alloc((size_t)NN * HH * 4);
  float*    embA   = (float*)alloc((size_t)TT * HH * 4);
  f16*      embA16 = (f16*)alloc((size_t)TT * HH * 2);
  float*    embB   = (float*)alloc((size_t)TT * HH * 4);
  f16*      embB16 = (f16*)alloc((size_t)TT * HH * 2);
  float*    qbuf   = (float*)alloc((size_t)TT * HH * 4);
  float*    kbuf   = (float*)alloc((size_t)TT * HH * 4);
  float*    vbuf   = (float*)alloc((size_t)TT * HH * 4);
  f16*      obuf16 = (f16*)alloc((size_t)TT * HH * 2);
  float*    proj   = (float*)alloc((size_t)TT * HH * 4);
  float*    x1     = (float*)alloc((size_t)TT * HH * 4);
  f16*      x1_16  = (f16*)alloc((size_t)TT * HH * 2);
  f16*      hid16  = (f16*)alloc((size_t)TT * FFH * 2);
  f16*      a1_16  = (f16*)alloc((size_t)TT * HH * 2);
  f16*      a2_16  = (f16*)alloc((size_t)TT * HH * 2);
  float*    wvals  = (float*)alloc((size_t)TT * 4);

  float* a1_out = (float*)d_out;                       // [3N,128]
  float* a2_out = a1_out + (size_t)TT * HH;            // [3N,128]
  float* g_out  = a2_out + (size_t)TT * HH;            // [N,128]

  // ---- helpers
  auto gemm = [&](const f16* A, const f16* Bt, const float* bias,
                  float* C, f16* C16, int M, int Nc, int K, int act) {
    dim3 g(M / 32, Nc / 128);
    if (act == 0)      gemm_wmma_f16<0><<<g, 128, 0, stream>>>(A, Bt, bias, C, C16, M, Nc, K);
    else if (act == 1) gemm_wmma_f16<1><<<g, 128, 0, stream>>>(A, Bt, bias, C, C16, M, Nc, K);
    else               gemm_wmma_f16<2><<<g, 128, 0, stream>>>(A, Bt, bias, C, C16, M, Nc, K);
  };
  auto transpose = [&](const float* W, f16* Wt, int K, int Nc) {
    long long n = (long long)K * Nc;
    k_w_transpose_f16<<<(unsigned)((n + 255) / 256), 256, 0, stream>>>(W, Wt, K, Nc);
  };
  auto zero = [&](void* ptr, long long words) {
    unsigned g = (unsigned)((words + 255) / 256); if (g > 4096) g = 4096;
    k_zero_u32<<<g, 256, 0, stream>>>((uint32_t*)ptr, words);
  };

  // ---- stage weights (f16, transposed) + f16 inputs
  for (int i = 0; i < VV; ++i) transpose(g1W[i], g1Wt[i], FEAT, HH);
  for (int i = 0; i < VV; ++i) transpose(g2W[i], g2Wt[i], HH, HH);
  for (int e = 0; e < 2; ++e) {
    transpose(encp[e][4], enc[e].Wqt, HH, HH);   // Wq
    transpose(encp[e][2], enc[e].Wkt, HH, HH);   // Wk
    transpose(encp[e][5], enc[e].Wvt, HH, HH);   // Wv
    transpose(encp[e][3], enc[e].Wot, HH, HH);   // Wo
    transpose(encp[e][0], enc[e].W1t, HH, FFH);  // W1
    transpose(encp[e][1], enc[e].W2t, FFH, HH);  // W2
  }
  transpose(attW1, attW1t, HH, HH);
  for (int i = 0; i < VV; ++i)
    k_f32_to_f16<<<4096, 256, 0, stream>>>(xin[i], x16[i], (long long)NN * FEAT);

  const int EB = (EE + NN + 255) / 256;        // edge kernels, thread/edge
  const int EW = (EE + NN + 7) / 8;            // edge aggregate, wave/edge

  auto run_gat = [&](const f16* A16, int K, f16* Wt,
                     const float* a_src, const float* a_dst, const float* bias,
                     const int* eidx, float* xpv, float* embF, f16* emb16) {
    gemm(A16, Wt, nullptr, xpv, nullptr, NN, HH, K, 0);
    k_rowdot2<<<(NN + 7) / 8, 256, 0, stream>>>(xpv, a_src, a_dst, a_s, a_d, NN);
    zero(maxb, NN); zero(denom, NN); zero(agg, (long long)NN * HH);
    k_edge_max<<<EB, 256, 0, stream>>>(eidx, eidx + EE, a_s, a_d, maxb);
    k_edge_exp<<<EB, 256, 0, stream>>>(eidx, eidx + EE, a_s, a_d, maxb, eexp, denom);
    k_edge_agg<<<EW, 256, 0, stream>>>(eidx, eidx + EE, eexp, denom, xpv, agg);
    k_gat_fin<<<(NN * HH) / 256, 256, 0, stream>>>(agg, bias, embF, emb16);
  };

  auto run_encoder = [&](const float* xf, const f16* xh, const EncDev& W,
                         float* outF, f16* out16) {
    gemm(xh, W.Wqt, W.bq, qbuf, nullptr, TT, HH, HH, 0);
    gemm(xh, W.Wkt, W.bk, kbuf, nullptr, TT, HH, HH, 0);
    gemm(xh, W.Wvt, W.bv, vbuf, nullptr, TT, HH, HH, 0);
    k_attn<<<(NN * 8) / 256, 256, 0, stream>>>(qbuf, kbuf, vbuf, obuf16);
    gemm(obuf16, W.Wot, W.bo, proj, nullptr, TT, HH, HH, 0);
    k_ln_res<<<TT / 8, 256, 0, stream>>>(proj, xf, W.ln1w, W.ln1b, x1, x1_16, TT);
    gemm(x1_16, W.W1t, W.b1, nullptr, hid16, TT, FFH, HH, 1);   // relu, f16-only out
    gemm(hid16, W.W2t, W.b2, proj, nullptr, TT, HH, FFH, 0);
    k_ln_res<<<TT / 8, 256, 0, stream>>>(proj, x1, W.ln2w, W.ln2b, outF, out16, TT);
  };

  // ---- pipeline
  for (int i = 0; i < VV; ++i)
    run_gat(x16[i], FEAT, g1Wt[i], g1as[i], g1ad[i], g1b[i], ei[i],
            xp + (size_t)i * NN * HH, embA + (size_t)i * NN * HH, embA16 + (size_t)i * NN * HH);
  run_encoder(embA, embA16, enc[0], a1_out, a1_16);

  for (int i = 0; i < VV; ++i)
    run_gat(a1_16 + (size_t)i * NN * HH, HH, g2Wt[i], g2as[i], g2ad[i], g2b[i], ei[i],
            xp + (size_t)i * NN * HH, embB + (size_t)i * NN * HH, embB16 + (size_t)i * NN * HH);
  run_encoder(embB, embB16, enc[1], a2_out, a2_16);

  // attention pooling: w = tanh(a2 @ W1 + b1) @ w2 ; beta = softmax over views
  gemm(a2_16, attW1t, attb1, qbuf /* reuse as tanh buf */, nullptr, TT, HH, HH, 2);
  k_rowdot2<<<TT / 8, 256, 0, stream>>>(qbuf, attw2, nullptr, wvals, nullptr, TT);
  k_pool<<<(NN * HH) / 256, 256, 0, stream>>>(wvals, a2_out, g_out);
}
